// TENSSDLayer_25082609008794
// MI455X (gfx1250) — matrix operand order repaired and checked by reading, not compiled, not run on hardware
//
#include <hip/hip_runtime.h>

typedef __bf16 bf16_t;
typedef __attribute__((ext_vector_type(16))) __bf16 v16bf;
typedef __attribute__((ext_vector_type(8)))  __bf16 v8bf;
typedef __attribute__((ext_vector_type(8)))  float  v8f;
typedef __attribute__((ext_vector_type(4)))  unsigned int u32x4;
typedef __attribute__((ext_vector_type(8)))  int i32x8;
typedef __attribute__((ext_vector_type(4)))  int i32x4;

#define BB   4
#define TT   4096
#define DD   1024
#define KKC  64
#define HHN  4
#define KH   16
#define C2K  128
#define MLPD 4096
#define BT   (BB*TT)

// GEMM tiling
#define MB        128
#define NB        128
#define KSTEP     64
#define LDS_STRIDE 72   // 64 bf16 + 8 bf16 (16B) pad per row -> conflict-free ds_load_b128

enum { EP_BIAS = 1, EP_SILU = 2, EP_SIG = 4, EP_MUL = 8, EP_ADD = 16 };

#if defined(__gfx1250__) && __has_builtin(__builtin_amdgcn_tensor_load_to_lds) && __has_builtin(__builtin_amdgcn_s_wait_tensorcnt)
#define HAVE_TDM 1
#else
#define HAVE_TDM 0
#endif

__device__ __forceinline__ bf16_t f2bf(float f) {
  unsigned u = __builtin_bit_cast(unsigned, f);
  unsigned r = (u + 0x7fffu + ((u >> 16) & 1u)) >> 16;
  unsigned short s = (unsigned short)r;
  return __builtin_bit_cast(bf16_t, s);
}

// ---------------- f32 -> bf16 convert ----------------
__global__ void cvt_bf16_kernel(const float* __restrict__ in, bf16_t* __restrict__ out, int n) {
  int i = blockIdx.x * 256 + threadIdx.x;
  if (i < n) out[i] = f2bf(in[i]);
}

// ---------------- LayerNorm (row per block), bf16 out ----------------
__global__ void ln_kernel(const float* __restrict__ x, const float* __restrict__ g,
                          const float* __restrict__ b, bf16_t* __restrict__ o) {
  __shared__ float s1[256], s2[256];
  int row = blockIdx.x;
  const float* xr = x + (size_t)row * DD;
  float v0 = xr[threadIdx.x], v1 = xr[threadIdx.x + 256];
  float v2 = xr[threadIdx.x + 512], v3 = xr[threadIdx.x + 768];
  s1[threadIdx.x] = v0 + v1 + v2 + v3;
  s2[threadIdx.x] = v0 * v0 + v1 * v1 + v2 * v2 + v3 * v3;
  __syncthreads();
  for (int s = 128; s > 0; s >>= 1) {
    if ((int)threadIdx.x < s) { s1[threadIdx.x] += s1[threadIdx.x + s]; s2[threadIdx.x] += s2[threadIdx.x + s]; }
    __syncthreads();
  }
  float m = s1[0] * (1.0f / DD);
  float var = s2[0] * (1.0f / DD) - m * m;
  float rinv = rsqrtf(var + 1e-5f);
  bf16_t* orow = o + (size_t)row * DD;
  float vv[4] = {v0, v1, v2, v3};
#pragma unroll
  for (int q = 0; q < 4; q++) {
    int col = threadIdx.x + q * 256;
    orow[col] = f2bf((vv[q] - m) * rinv * g[col] + b[col]);
  }
}

// ---------------- TDM tile load: 2D tile (KSTEP x MB rows) of bf16, padded rows ----------------
__device__ __forceinline__ void tdm_issue_tile(const bf16_t* gtile, int kDim, int rowsDim,
                                               unsigned lds_off) {
#if HAVE_TDM
  unsigned long long ga = (unsigned long long)(uintptr_t)gtile;
  u32x4 g0;
  g0[0] = 1u;                                   // count=1, user mode, no gather
  g0[1] = lds_off;                              // LDS byte address
  g0[2] = (unsigned)(ga & 0xffffffffull);       // global_addr[31:0]
  g0[3] = (unsigned)((ga >> 32) & 0x01ffffffull) | (2u << 30);  // addr[56:32] | type=2
  i32x8 g1;
  // data_size=2B(code1) | pad_enable | pad_interval=32dw(code4) | pad_amount=4dw(code3)
  g1[0] = (int)((1u << 16) | (1u << 20) | (4u << 22) | (3u << 25));
  g1[1] = (int)(((unsigned)kDim & 0xffffu) << 16);              // tensor_dim0 lo
  g1[2] = (int)((((unsigned)kDim >> 16) & 0xffffu) |
                (((unsigned)rowsDim & 0xffffu) << 16));         // dim0 hi | dim1 lo
  g1[3] = (int)((((unsigned)rowsDim >> 16) & 0xffffu) |
                ((unsigned)KSTEP << 16));                       // dim1 hi | tile_dim0
  g1[4] = (int)(unsigned)MB;                                    // tile_dim1 | tile_dim2=0
  g1[5] = kDim;                                                 // tensor_dim0_stride lo
  g1[6] = 0;                                                    // stride0 hi | stride1 lo
  g1[7] = 0;
  i32x4 gz4 = {0, 0, 0, 0};
#if __clang_major__ >= 23
  i32x8 gz8 = {0, 0, 0, 0, 0, 0, 0, 0};
  __builtin_amdgcn_tensor_load_to_lds(g0, g1, gz4, gz4, gz8, 0);
#else
  __builtin_amdgcn_tensor_load_to_lds(g0, g1, gz4, gz4, 0);
#endif
#else
  (void)gtile; (void)kDim; (void)rowsDim; (void)lds_off;
#endif
}

// fallback cooperative stage (one row of 64 bf16 per thread, padded LDS layout)
__device__ __forceinline__ void coop_load_tile(const bf16_t* __restrict__ g, int ldg,
                                               bf16_t* __restrict__ lds, int tid) {
  const bf16_t* src = g + (size_t)tid * ldg;
  bf16_t* dst = lds + tid * LDS_STRIDE;
#pragma unroll
  for (int q = 0; q < 8; q++)
    *(v8bf*)(dst + q * 8) = *(const v8bf*)(src + q * 8);
}

// ---------------- WMMA bf16 GEMM: C[M,N] = A[M,K] @ W[N,K]^T + fused epilogue ----------------
// 128 threads = 4 waves (2x2), block tile 128x128, K step 64, TDM double-buffered LDS.
__global__ void __launch_bounds__(128) gemm_bf16_wmma(
    const bf16_t* __restrict__ A, const bf16_t* __restrict__ W,
    int M, int N, int K,
    const float* __restrict__ bias, const float* __restrict__ mulsrc,
    const float* __restrict__ addsrc, float* __restrict__ outF,
    bf16_t* __restrict__ outH, int mode) {
  __shared__ bf16_t As[2][MB * LDS_STRIDE];
  __shared__ bf16_t Bs[2][NB * LDS_STRIDE];

  int tiles_n = N >> 7;
  int tm = (blockIdx.x / tiles_n) << 7;
  int tn = (blockIdx.x % tiles_n) << 7;
  int lane = threadIdx.x & 31;
  int wave = threadIdx.x >> 5;
  int wm = (wave >> 1) * 64;
  int wn = (wave & 1) * 64;
  int hf = lane >> 4;
  int l16 = lane & 15;

  const bf16_t* aTile = A + (size_t)tm * K;
  const bf16_t* wTile = W + (size_t)tn * K;

  v8f zero = {0.f, 0.f, 0.f, 0.f, 0.f, 0.f, 0.f, 0.f};
  v8f acc[4][4];
#pragma unroll
  for (int i = 0; i < 4; i++)
#pragma unroll
    for (int j = 0; j < 4; j++) acc[i][j] = zero;

  int nsteps = K / KSTEP;

  // prologue: fill buffer 0
#if HAVE_TDM
  if (wave == 0) {
    tdm_issue_tile(aTile, K, M, (unsigned)(unsigned long long)(uintptr_t)&As[0][0]);
    tdm_issue_tile(wTile, K, N, (unsigned)(unsigned long long)(uintptr_t)&Bs[0][0]);
    __builtin_amdgcn_s_wait_tensorcnt((short)0);
  }
#else
  coop_load_tile(aTile, K, &As[0][0], threadIdx.x);
  coop_load_tile(wTile, K, &Bs[0][0], threadIdx.x);
#endif
  __syncthreads();

  for (int s = 0; s < nsteps; s++) {
    int cur = s & 1;
    int nxt = cur ^ 1;
    bool more = (s + 1) < nsteps;
    if (more) {
      const bf16_t* aN = aTile + (size_t)(s + 1) * KSTEP;
      const bf16_t* wN = wTile + (size_t)(s + 1) * KSTEP;
#if HAVE_TDM
      if (wave == 0) {
        tdm_issue_tile(aN, K, M, (unsigned)(unsigned long long)(uintptr_t)&As[nxt][0]);
        tdm_issue_tile(wN, K, N, (unsigned)(unsigned long long)(uintptr_t)&Bs[nxt][0]);
      }
#else
      coop_load_tile(aN, K, &As[nxt][0], threadIdx.x);
      coop_load_tile(wN, K, &Bs[nxt][0], threadIdx.x);
#endif
    }
    // compute: two 32-deep sub-steps from the current LDS tile
#pragma unroll
    for (int sub = 0; sub < 2; sub++) {
      v16bf af[4], bfr[4];
#pragma unroll
      for (int i = 0; i < 4; i++) {
        const bf16_t* p = &As[cur][(wm + i * 16 + l16) * LDS_STRIDE + sub * 32 + hf * 8];
        union { v16bf v; v8bf h[2]; } u;
        u.h[0] = *(const v8bf*)p;
        u.h[1] = *(const v8bf*)(p + 16);
        af[i] = u.v;
      }
#pragma unroll
      for (int j = 0; j < 4; j++) {
        const bf16_t* p = &Bs[cur][(wn + j * 16 + l16) * LDS_STRIDE + sub * 32 + hf * 8];
        union { v16bf v; v8bf h[2]; } u;
        u.h[0] = *(const v8bf*)p;
        u.h[1] = *(const v8bf*)(p + 16);
        bfr[j] = u.v;
      }
#pragma unroll
      for (int i = 0; i < 4; i++)
#pragma unroll
        for (int j = 0; j < 4; j++)
          acc[i][j] = __builtin_amdgcn_wmma_f32_16x16x32_bf16(
              false, af[i], false, bfr[j], (short)0, acc[i][j], false, false);
    }
    if (more) {
#if HAVE_TDM
      if (wave == 0) __builtin_amdgcn_s_wait_tensorcnt((short)0);
#endif
      __syncthreads();
    }
  }

  // Epilogue. C layout: lanes 0-15 -> M = r, lanes 16-31 -> M = 8+r; N = l16.
#pragma unroll
  for (int i = 0; i < 4; i++) {
#pragma unroll
    for (int j = 0; j < 4; j++) {
      int col = tn + wn + j * 16 + l16;
#pragma unroll
      for (int r = 0; r < 8; r++) {
        int row = tm + wm + i * 16 + hf * 8 + r;
        size_t idx = (size_t)row * N + col;
        float v = acc[i][j][r];
        if (mode & EP_BIAS) v += bias[col];
        if (mode & EP_SILU) v = v / (1.0f + __expf(-v));
        if (mode & EP_SIG)  v = 1.0f / (1.0f + __expf(-v));
        if (mode & EP_MUL)  v *= mulsrc[idx];
        if (mode & EP_ADD)  v += addsrc[idx];
        if (outF) outF[idx] = v;
        if (outH) outH[idx] = f2bf(v);
      }
    }
  }
}

// ---------------- depthwise causal conv (k=4) + bias + SiLU ----------------
__global__ void conv_silu_kernel(const float* __restrict__ beta, const float* __restrict__ cw,
                                 const float* __restrict__ cb, float* __restrict__ out) {
  int i = blockIdx.x * 256 + threadIdx.x;
  if (i >= BT * C2K) return;
  int c = i & (C2K - 1);
  int bt = i >> 7;
  int t = bt & (TT - 1);
  float v = cb[c];
#pragma unroll
  for (int j = 0; j < 4; j++) {
    int tt2 = t - 3 + j;
    if (tt2 >= 0) v += cw[c * 4 + j] * beta[(size_t)(bt - 3 + j) * C2K + c];
  }
  out[i] = v / (1.0f + __expf(-v));
}

// ---------------- SSD: complex linear recurrence matching the reference's
// chunk-boundary semantics (carry crosses each 64-boundary with multiplier 1).
__global__ void ssd_scan_kernel(const float* __restrict__ bsrc, float* __restrict__ cdst,
                                const float* __restrict__ log_decay, const float* __restrict__ freq) {
  int b = blockIdx.x >> 6;
  int k = blockIdx.x & 63;
  int lane = threadIdx.x;
  float mag = 1.0f / (1.0f + __expf(-log_decay[k]));
  float fr = freq[k];
  float ar = mag * __cosf(fr), ai = mag * __sinf(fr);
  const float* br = bsrc + (size_t)b * TT * C2K + k;
  const float* bi = br + KKC;
  int t0 = lane * 128;

  float sr = 0.f, si = 0.f;
  for (int j = 0; j < 128; j++) {
    int t = t0 + j;
    float xr = br[(size_t)t * C2K];
    float xi = bi[(size_t)t * C2K];
    if ((t & 63) == 0) { sr += xr; si += xi; }
    else { float nr = ar * sr - ai * si + xr; float ni = ar * si + ai * sr + xi; sr = nr; si = ni; }
  }
  float dr = 1.f, di = 0.f;
  for (int q = 0; q < 126; q++) { float nr = dr * ar - di * ai; float ni = dr * ai + di * ar; dr = nr; di = ni; }
  for (int off = 1; off < 32; off <<= 1) {
    float psr = __shfl_up(sr, off, 32);
    float psi = __shfl_up(si, off, 32);
    float pdr = __shfl_up(dr, off, 32);
    float pdi = __shfl_up(di, off, 32);
    if (lane >= off) {
      float nsr = sr + dr * psr - di * psi;
      float nsi = si + dr * psi + di * psr;
      float ndr = dr * pdr - di * pdi;
      float ndi = dr * pdi + di * pdr;
      sr = nsr; si = nsi; dr = ndr; di = ndi;
    }
  }
  float cr = __shfl_up(sr, 1, 32);
  float ci = __shfl_up(si, 1, 32);
  if (lane == 0) { cr = 0.f; ci = 0.f; }
  sr = cr; si = ci;
  float* orp = cdst + (size_t)b * TT * C2K + k;
  float* oip = orp + KKC;
  for (int j = 0; j < 128; j++) {
    int t = t0 + j;
    float xr = br[(size_t)t * C2K];
    float xi = bi[(size_t)t * C2K];
    if ((t & 63) == 0) { sr += xr; si += xi; }
    else { float nr = ar * sr - ai * si + xr; float ni = ar * si + ai * sr + xi; sr = nr; si = ni; }
    orp[(size_t)t * C2K] = sr;
    oip[(size_t)t * C2K] = si;
  }
}

// ---------------- per-head coupling ----------------
__global__ void coupling_kernel(const float* __restrict__ c, const float* __restrict__ coup,
                                float* __restrict__ eigF, bf16_t* __restrict__ eigH) {
  __shared__ float rowv[C2K];
  __shared__ float cp[HHN * KH * KH];
  int bt = blockIdx.x;
  int ch = threadIdx.x;
  rowv[ch] = c[(size_t)bt * C2K + ch];
  for (int q = ch; q < HHN * KH * KH; q += C2K) cp[q] = coup[q];
  __syncthreads();
  int part = ch >> 6;
  int r6 = ch & 63;
  int h = r6 >> 4;
  int j = r6 & 15;
  const float* cw2 = cp + (h * KH + j) * KH;
  const float* rv = rowv + part * KKC + h * KH;
  float v = 0.f;
#pragma unroll
  for (int k = 0; k < KH; k++) v += cw2[k] * rv[k];
  eigF[(size_t)bt * C2K + ch] = v;
  eigH[(size_t)bt * C2K + ch] = f2bf(v);
}

extern "C" void kernel_launch(void* const* d_in, const int* in_sizes, int n_in,
                              void* d_out, int out_size, void* d_ws, size_t ws_size,
                              hipStream_t stream) {
  (void)in_sizes; (void)n_in; (void)out_size; (void)ws_size;
  const float* x       = (const float*)d_in[0];
  const float* in_w    = (const float*)d_in[1];
  const float* conv_w  = (const float*)d_in[2];
  const float* conv_b  = (const float*)d_in[3];
  const float* log_dec = (const float*)d_in[4];
  const float* freq    = (const float*)d_in[5];
  const float* coup    = (const float*)d_in[6];
  const float* out_w   = (const float*)d_in[7];
  const float* gate_w  = (const float*)d_in[8];
  const float* gate_b  = (const float*)d_in[9];
  const float* m1_w    = (const float*)d_in[10];
  const float* m1_b    = (const float*)d_in[11];
  const float* m2_w    = (const float*)d_in[12];
  const float* m2_b    = (const float*)d_in[13];
  const float* ln1_g   = (const float*)d_in[14];
  const float* ln1_b   = (const float*)d_in[15];
  const float* ln2_g   = (const float*)d_in[16];
  const float* ln2_b   = (const float*)d_in[17];

  float* outp = (float*)d_out;
  float* eigF = outp + (size_t)BT * DD;

  char* wsb = (char*)d_ws;
  size_t off = 0;
  auto alloc = [&](size_t bytes) -> char* {
    char* p = wsb + off;
    off += (bytes + 255) & ~(size_t)255;
    return p;
  };
  bf16_t* xn_bf    = (bf16_t*)alloc((size_t)BT * DD * 2);
  bf16_t* w_in_bf  = (bf16_t*)alloc((size_t)C2K * DD * 2);
  bf16_t* w_out_bf = (bf16_t*)alloc((size_t)DD * C2K * 2);
  bf16_t* w_gt_bf  = (bf16_t*)alloc((size_t)DD * DD * 2);
  bf16_t* w_m1_bf  = (bf16_t*)alloc((size_t)MLPD * DD * 2);
  bf16_t* w_m2_bf  = (bf16_t*)alloc((size_t)DD * MLPD * 2);
  float*  beta     = (float*)alloc((size_t)BT * C2K * 4);
  float*  betac    = (float*)alloc((size_t)BT * C2K * 4);
  float*  cbuf     = (float*)alloc((size_t)BT * C2K * 4);
  bf16_t* eig_bf   = (bf16_t*)alloc((size_t)BT * C2K * 2);
  float*  hbuf     = (float*)alloc((size_t)BT * DD * 4);
  float*  x1       = (float*)alloc((size_t)BT * DD * 4);
  bf16_t* xn2_bf   = (bf16_t*)alloc((size_t)BT * DD * 2);
  bf16_t* y1_bf    = (bf16_t*)alloc((size_t)BT * MLPD * 2);

  auto cvt = [&](const float* src, bf16_t* dst, int n) {
    cvt_bf16_kernel<<<(n + 255) / 256, 256, 0, stream>>>(src, dst, n);
  };
  cvt(in_w,   w_in_bf,  C2K * DD);
  cvt(out_w,  w_out_bf, DD * C2K);
  cvt(gate_w, w_gt_bf,  DD * DD);
  cvt(m1_w,   w_m1_bf,  MLPD * DD);
  cvt(m2_w,   w_m2_bf,  DD * MLPD);

  // LN1: x -> xn (bf16)
  ln_kernel<<<BT, 256, 0, stream>>>(x, ln1_g, ln1_b, xn_bf);

  // beta = xn @ in_proj_w.T    (M=BT, N=128, K=1024)
  gemm_bf16_wmma<<<(BT / MB) * (C2K / NB), 128, 0, stream>>>(
      xn_bf, w_in_bf, BT, C2K, DD, nullptr, nullptr, nullptr, beta, nullptr, 0);

  conv_silu_kernel<<<(BT * C2K + 255) / 256, 256, 0, stream>>>(beta, conv_w, conv_b, betac);

  ssd_scan_kernel<<<BB * KKC, 32, 0, stream>>>(betac, cbuf, log_dec, freq);

  coupling_kernel<<<BT, C2K, 0, stream>>>(cbuf, coup, eigF, eig_bf);

  // h = eig @ out_proj_w.T     (M=BT, N=1024, K=128)
  gemm_bf16_wmma<<<(BT / MB) * (DD / NB), 128, 0, stream>>>(
      eig_bf, w_out_bf, BT, DD, C2K, nullptr, nullptr, nullptr, hbuf, nullptr, 0);

  // x1 = x + sigmoid(xn @ gate_w.T + gate_b) * h
  gemm_bf16_wmma<<<(BT / MB) * (DD / NB), 128, 0, stream>>>(
      xn_bf, w_gt_bf, BT, DD, DD, gate_b, hbuf, x, x1, nullptr,
      EP_BIAS | EP_SIG | EP_MUL | EP_ADD);

  ln_kernel<<<BT, 256, 0, stream>>>(x1, ln2_g, ln2_b, xn2_bf);

  // y1 = silu(xn2 @ mlp_w1.T + b1) -> bf16 directly
  gemm_bf16_wmma<<<(BT / MB) * (MLPD / NB), 128, 0, stream>>>(
      xn2_bf, w_m1_bf, BT, MLPD, DD, m1_b, nullptr, nullptr, nullptr, y1_bf,
      EP_BIAS | EP_SILU);

  // out = x1 + y1 @ mlp_w2.T + b2   (directly into d_out)
  gemm_bf16_wmma<<<(BT / MB) * (DD / NB), 128, 0, stream>>>(
      y1_bf, w_m2_bf, BT, DD, MLPD, m2_b, nullptr, x1, outp, nullptr,
      EP_BIAS | EP_ADD);
}
